// MultiEncoding__32993938767928
// MI455X (gfx1250) — matrix operand order repaired and checked by reading, not compiled, not run on hardware
//
#include <hip/hip_runtime.h>
#include <hip/hip_bf16.h>
#include <math.h>

typedef __attribute__((ext_vector_type(16))) _Float16 v16h;
typedef __attribute__((ext_vector_type(8)))  float    v8f;

#define NPOS  13
#define NSTEP 169          // 13*13
#define DIN   676          // 169*4
#define NG    22           // K groups of 32 (KPAD = 704)
#define RSQRT2 0.70710678118654752440f

// fragA layout: [NG][T][32 lanes][16 elems] f16, T = B/16 row tiles.
// Per-lane 16 f16 = 32 contiguous bytes = exactly one WMMA A fragment slice.

// ---------------------------------------------------------------------------
// Kernel 0: zero the last K-group block of fragA (covers k = 672..703; the
// sim later overwrites the valid k = 672..675 entries; 676..703 stay zero).
// ---------------------------------------------------------------------------
__global__ void pad_zero_kernel(_Float16* __restrict__ fragA, int T) {
    int t = blockIdx.x * blockDim.x + threadIdx.x;
    int total = T * 32 * 16;                       // one group block
    if (t < total) fragA[(size_t)(NG - 1) * total + t] = (_Float16)0.0f;
}

// ---------------------------------------------------------------------------
// Kernel 1: pack W1 (f32 [20][676]) into WMMA B fragments (one-shot, 45 KB).
// fragB layout: [NG][2 ntiles][32 lanes][16 elems] f16.
// B[k][n] = W1[n][k]; lane = half*16 + n%16; elem e -> k = g*32 + 8*half + (e&7) + ((e>>3)<<4).
// ---------------------------------------------------------------------------
__global__ void pack_w1_kernel(const float* __restrict__ W1,
                               _Float16* __restrict__ fragB) {
    int t = blockIdx.x * blockDim.x + threadIdx.x;
    if (t >= NG * 2 * 32 * 16) return;
    const int e    = t & 15;
    const int lane = (t >> 4) & 31;
    const int nt   = (t >> 9) & 1;
    const int g    = t >> 10;
    const int half = lane >> 4, l16 = lane & 15;
    const int n = nt * 16 + l16;
    const int k = g * 32 + half * 8 + (e & 7) + ((e >> 3) << 4);
    float w = (n < 20 && k < DIN) ? W1[n * DIN + k] : 0.0f;
    fragB[t] = (_Float16)w;
}

// ---------------------------------------------------------------------------
// Kernel 2: quantum filter chain.
// lane = half*16 + ampIdx; each wave32 simulates TWO batch elements.
// Gate pair-exchange via __shfl_xor (LDS permute path); 1 sincos per lane
// per step; angle broadcast via __shfl in the 16-lane group.
// Measured features are scattered directly into WMMA A-fragment layout.
// ---------------------------------------------------------------------------
__global__ void __launch_bounds__(256)
qsim_kernel(const float* __restrict__ x,       // [B][784]
            const float* __restrict__ crz_p,   // scalar
            const float* __restrict__ ry_p,    // scalar
            _Float16* __restrict__ fragA,      // [NG][T][32][16]
            int B) {
    const int tid  = blockIdx.x * blockDim.x + threadIdx.x;
    const int lane = threadIdx.x & 31;
    const int half = lane >> 4;
    const int ai   = lane & 15;          // amplitude index 0..15
    const int b    = (tid >> 5) * 2 + half;
    if (b >= B) return;
    const int T = B >> 4;

    // qubit q <-> bit (3-q) of amplitude index (reference axis order)
    int bbit[4];
    float sq[4];   // RZ/RY sigma sign: bit ? +1 : -1
    float hs[4];   // H own-sign:       bit ? -1 : +1
#pragma unroll
    for (int q = 0; q < 4; ++q) {
        bbit[q] = (ai >> (3 - q)) & 1;
        sq[q]   = bbit[q] ? 1.0f : -1.0f;
        hs[q]   = bbit[q] ? -1.0f : 1.0f;
    }

    const float crz = crz_p[0] * 0.5f;
    const float ryt = ry_p[0] * 0.5f;
    float ss, cc, sr, cr;
    sincosf(crz, &ss, &cc);
    sincosf(ryt, &sr, &cr);

    float re = (ai == 0) ? 1.0f : 0.0f;
    float im = 0.0f;

    const float* __restrict__ xb = x + (size_t)b * 784;
    const int pr = ai >> 2, pc = ai & 3;

    // destination slot (fragA) pieces that depend only on b
    const int tile = b >> 4, l16b = b & 15;

    const int g_rz[16] = {1,0,1,0,1, 1,0,1,0,1, 1,0,1,0,1, 1};
    const int g_q [16] = {0,0,0,0,0, 1,1,1,1,1, 2,2,2,2,2, 3};

    for (int p = 0; p < NSTEP; ++p) {
        const int py = p / NPOS, px = p - py * NPOS;
        const float f = xb[(py * 2 + pr) * 28 + (px * 2 + pc)];
        // speculative prefetch of next patch row (fault-free per ISA)
        {
            const int pn = (p + 1 < NSTEP) ? p + 1 : p;
            const int pyn = pn / NPOS, pxn = pn - pyn * NPOS;
            __builtin_prefetch(&xb[(pyn * 2 + pr) * 28 + (pxn * 2 + pc)], 0, 3);
        }
        float se, ce;
        sincosf(0.5f * f, &se, &ce);

        // ---- Hadamard on all 4 qubits ----
#pragma unroll
        for (int q = 0; q < 4; ++q) {
            const int msk = 8 >> q;
            float pre = __shfl_xor(re, msk);
            float pim = __shfl_xor(im, msk);
            re = (pre + hs[q] * re) * RSQRT2;
            im = (pim + hs[q] * im) * RSQRT2;
        }

        // ---- 16 data-encoding gates ----
#pragma unroll
        for (int g = 0; g < 16; ++g) {
            const int q = g_q[g];
            const float c = __shfl(ce, half * 16 + g, 32);
            const float s = __shfl(se, half * 16 + g, 32);
            const float s2 = sq[q] * s;
            if (g_rz[g]) {               // RZ: diag(e^{-i s}, e^{+i s})
                float nr = re * c - im * s2;
                float ni = re * s2 + im * c;
                re = nr; im = ni;
            } else {                     // RY: [[c,-s],[s,c]]
                const int msk = 8 >> q;
                float pre = __shfl_xor(re, msk);
                float pim = __shfl_xor(im, msk);
                re = c * re + s2 * pre;
                im = c * im + s2 * pim;
            }
        }

        // ---- CRZ ring (ctrl i -> tgt (i+1)%4), diagonal phase ----
#pragma unroll
        for (int cdx = 0; cdx < 4; ++cdx) {
            const int tdx = (cdx + 1) & 3;
            const float rho = bbit[cdx] ? cc : 1.0f;
            const float sg  = bbit[cdx] ? (sq[tdx] * ss) : 0.0f;
            float nr = re * rho - im * sg;
            float ni = re * sg + im * rho;
            re = nr; im = ni;
        }

        // ---- shared RY on all 4 qubits ----
#pragma unroll
        for (int q = 0; q < 4; ++q) {
            const int msk = 8 >> q;
            float pre = __shfl_xor(re, msk);
            float pim = __shfl_xor(im, msk);
            const float s2 = sq[q] * sr;
            re = cr * re + s2 * pre;
            im = cr * im + s2 * pim;
        }

        // ---- measure <Z_q>, scatter into WMMA A-fragment layout ----
        const float prob = re * re + im * im;
#pragma unroll
        for (int q = 0; q < 4; ++q) {
            float v = bbit[q] ? -prob : prob;
            v += __shfl_xor(v, 1);
            v += __shfl_xor(v, 2);
            v += __shfl_xor(v, 4);
            v += __shfl_xor(v, 8);
            if (ai == q) {
                const int k  = p * 4 + q;
                const int g  = k >> 5, kr = k & 31;
                const int hh = (kr >> 3) & 1;               // lane half slot
                const int e  = (kr & 7) + ((kr >> 4) << 3); // elem slot
                const size_t idx =
                    ((((size_t)g * T + tile) * 32) + hh * 16 + l16b) * 16 + e;
                fragA[idx] = (_Float16)v;
            }
        }
    }
}

// ---------------------------------------------------------------------------
// Kernel 3: MLP head via V_WMMA_F32_16X16X32_F16 on pre-packed fragments.
// One wave32 per 16-row tile; inner loop: 3 contiguous 32B loads + 2 WMMAs.
// ---------------------------------------------------------------------------
__global__ void __launch_bounds__(32)
mlp_wmma_kernel(const _Float16* __restrict__ fragA,  // [NG][T][32][16]
                const _Float16* __restrict__ fragB,  // [NG][2][32][16]
                const float* __restrict__ bias1,     // [20]
                const float* __restrict__ W2,        // [2][20]
                const float* __restrict__ bias2,     // [2]
                float* __restrict__ out,             // [B][2]
                int T) {
    const int lane = threadIdx.x;
    const int half = lane >> 4;
    const int l16  = lane & 15;
    const int tile = blockIdx.x;

    v8f acc0 = {}; v8f acc1 = {};

#pragma unroll 2
    for (int g = 0; g < NG; ++g) {
        const v16h a  = *(const v16h*)(fragA +
                          ((((size_t)g * T + tile) * 32) + lane) * 16);
        const v16h b0 = *(const v16h*)(fragB + (((size_t)g * 2 + 0) * 32 + lane) * 16);
        const v16h b1 = *(const v16h*)(fragB + (((size_t)g * 2 + 1) * 32 + lane) * 16);
        acc0 = __builtin_amdgcn_wmma_f32_16x16x32_f16(
                   false, a, false, b0, (short)0, acc0, false, false);
        acc1 = __builtin_amdgcn_wmma_f32_16x16x32_f16(
                   false, a, false, b1, (short)0, acc1, false, false);
    }

    // C layout: VGPR r -> M = r + 8*half, N = lane%16
    __shared__ float h[16][20];
#pragma unroll
    for (int r = 0; r < 8; ++r) {
        const int m = r + 8 * half;
        const int n = l16;
        float v = acc0[r] + bias1[n];
        h[m][n] = (v > 0.0f) ? v : 0.1f * v;
        const int n2 = 16 + l16;
        if (n2 < 20) {
            float v2 = acc1[r] + bias1[n2];
            h[m][n2] = (v2 > 0.0f) ? v2 : 0.1f * v2;
        }
    }
    __syncthreads();

    if (lane < 16) {
        const int m = lane;
        float o0 = bias2[0], o1 = bias2[1];
#pragma unroll
        for (int n = 0; n < 20; ++n) {
            const float hv = h[m][n];
            o0 += hv * W2[n];
            o1 += hv * W2[20 + n];
        }
        const int brow = tile * 16 + m;
        out[brow * 2 + 0] = o0;
        out[brow * 2 + 1] = o1;
    }
}

// ---------------------------------------------------------------------------
extern "C" void kernel_launch(void* const* d_in, const int* in_sizes, int n_in,
                              void* d_out, int out_size, void* d_ws, size_t ws_size,
                              hipStream_t stream) {
    (void)n_in; (void)out_size; (void)ws_size;
    const float* x    = (const float*)d_in[0];   // [B,1,28,28]
    const float* crz  = (const float*)d_in[1];   // scalar
    const float* ryt  = (const float*)d_in[2];   // scalar
    const float* W1   = (const float*)d_in[3];   // [20,676]
    const float* b1   = (const float*)d_in[4];   // [20]
    const float* W2   = (const float*)d_in[5];   // [2,20]
    const float* b2   = (const float*)d_in[6];   // [2]
    float*       out  = (float*)d_out;           // [B,2]

    const int B = in_sizes[0] / 784;             // 8192
    const int T = B / 16;                        // 512 row tiles

    _Float16* fragA = (_Float16*)d_ws;                     // NG*T*32*16 f16 ~11.5MB
    _Float16* fragB = fragA + (size_t)NG * T * 32 * 16;    // NG*2*32*16 f16 = 45KB

    // 1) zero last-K-group fragA block (pad k=676..703)
    const int padElems = T * 32 * 16;
    pad_zero_kernel<<<(padElems + 255) / 256, 256, 0, stream>>>(fragA, T);

    // 2) pack W1 into B fragments (one-shot)
    const int wElems = NG * 2 * 32 * 16;
    pack_w1_kernel<<<(wElems + 255) / 256, 256, 0, stream>>>(W1, fragB);

    // 3) quantum sim: 16 lanes per batch element
    const int simThreads = B * 16;
    qsim_kernel<<<simThreads / 256, 256, 0, stream>>>(x, crz, ryt, fragA, B);

    // 4) WMMA MLP head: one wave per 16 batch rows
    mlp_wmma_kernel<<<T, 32, 0, stream>>>(fragA, fragB, b1, W2, b2, out, T);
}
